// NLinear_2113123910010
// MI455X (gfx1250) — matrix operand order repaired and checked by reading, not compiled, not run on hardware
//
#include <hip/hip_runtime.h>

typedef __attribute__((ext_vector_type(16))) _Float16 v16h;
typedef __attribute__((ext_vector_type(8)))  float    v8f;

constexpr int Bn = 4096;
constexpr int Sn = 512;
constexpr int Cn = 64;
constexpr int Pn = 336;

// Workgroup tile: 32 b-rows x 48 p-cols x 16 channels, K-step = 32.
// 256 threads = 8 waves; wave w owns channels {2w, 2w+1}; each wave holds
// 2ch x 2M x 3N = 12 f32 16x16 accumulators (96 VGPRs).
//
// LDS (160 KB -> 2 WGs/WGP so staging of one WG overlaps compute of the other):
//   Ah/Al : [16ch][2mt][32lane][16 f16]  = 32 KB each  (x hi/lo fragments)
//   Bh/Bl : [16ch][3nt][32lane][16 f16]  = 48 KB each  (W hi/lo fragments)
// Epilogue aliases the whole buffer as f32 [32m][48p][17] (pad avoids bank
// conflicts) to assemble c-contiguous coalesced stores.

__global__ __launch_bounds__(256)
void nlinear_wmma_f32split(const float* __restrict__ x,
                           const float* __restrict__ w,
                           const float* __restrict__ bias,
                           float* __restrict__ out)
{
    __shared__ __align__(32) unsigned char smem[163840];
    _Float16* Ah = (_Float16*)(smem);
    _Float16* Al = (_Float16*)(smem + 32768);
    _Float16* Bh = (_Float16*)(smem + 65536);
    _Float16* Bl = (_Float16*)(smem + 114688);
    float*    Fo = (float*)(smem);              // epilogue alias [32][48][17]

    const int tid  = threadIdx.x;
    const int lane = tid & 31;
    const int wave = tid >> 5;

    const int p0 = blockIdx.x * 48;
    const int b0 = blockIdx.y * 32;
    const int c0 = blockIdx.z * 16;

    // staging mapping: thread -> channel cA (coalesced along c) and rows {2q,2q+1}
    const int cA = tid & 15;
    const int q  = tid >> 4;

    // seq_last values this thread needs (same (row, channel) pairs as staging/store)
    const float sl0 = x[(size_t)(b0 + 2*q    ) * (Sn*Cn) + (size_t)(Sn-1)*Cn + c0 + cA];
    const float sl1 = x[(size_t)(b0 + 2*q + 1) * (Sn*Cn) + (size_t)(Sn-1)*Cn + c0 + cA];

    v8f acc[2][2][3];
#pragma unroll
    for (int a = 0; a < 2; ++a)
#pragma unroll
        for (int m = 0; m < 2; ++m)
#pragma unroll
            for (int n = 0; n < 3; ++n)
                acc[a][m][n] = (v8f){0.f,0.f,0.f,0.f,0.f,0.f,0.f,0.f};

    for (int ks = 0; ks < Sn / 32; ++ks) {
        const int s0 = ks * 32;

        // ---- stage X: subtract seq_last, split f32 -> (hi,lo) f16, scatter into
        //      the ISA A-matrix (16x32 f16) per-lane layout ----
#pragma unroll
        for (int j = 0; j < 2; ++j) {
            const int   i   = 2*q + j;
            const float slv = j ? sl1 : sl0;
            const float* xp = x + (size_t)(b0 + i) * (Sn*Cn) + (size_t)s0 * Cn + c0 + cA;
            const int m  = i & 15;
            const int mt = i >> 4;
#pragma unroll
            for (int s = 0; s < 32; ++s) {
                float v = xp[s * Cn] - slv;
                _Float16 h = (_Float16)v;
                _Float16 l = (_Float16)(v - (float)h);
                const int lt  = m + (((s >> 3) & 1) << 4);   // lane
                const int el  = (s & 7) + ((s >> 4) << 3);   // half index in v16h
                const int idx = (((cA*2 + mt) * 32) + lt) * 16 + el;
                Ah[idx] = h;
                Al[idx] = l;
            }
        }

        // ---- stage W: split into hi/lo, scatter into B-matrix (32x16 f16) layout ----
#pragma unroll 8
        for (int r = 0; r < 96; ++r) {
            const int f   = r * 256 + tid;           // p runs fastest across threads
            const int ch  = f / 1536;                // 32*48
            const int rem = f - ch * 1536;
            const int sj  = rem / 48;
            const int pk  = rem - sj * 48;
            float wv = w[(size_t)(c0 + ch) * (Sn*Pn) + (size_t)(s0 + sj) * Pn + p0 + pk];
            _Float16 h = (_Float16)wv;
            _Float16 l = (_Float16)(wv - (float)h);
            const int nt  = pk >> 4;
            const int n   = pk & 15;
            const int lt  = n + (((sj >> 3) & 1) << 4);    // lane: lower half K=0..7,16..23, upper K=8..15,24..31
            const int el  = (sj & 7) + ((sj >> 4) << 3);
            const int idx = ((ch*3 + nt) * 32 + lt) * 16 + el;
            Bh[idx] = h;
            Bl[idx] = l;
        }
        __syncthreads();

        // prefetch next K-step while WMMAs run
        if (ks + 1 < Sn / 32) {
            __builtin_prefetch(x + (size_t)(b0 + 2*q) * (Sn*Cn) + (size_t)(s0 + 32) * Cn + c0 + cA, 0, 3);
            __builtin_prefetch(w + (size_t)(c0 + (tid >> 4)) * (Sn*Pn) + (size_t)(s0 + 32 + (tid & 15)) * Pn + p0, 0, 3);
        }

        // ---- compute: per channel, hoist all fragments to registers, then
        //      2M x 3N tiles * 3 WMMAs (split-f16 fp32 emulation) ----
#pragma unroll
        for (int c2 = 0; c2 < 2; ++c2) {
            const int ch = wave * 2 + c2;
            v16h a_h[2], a_l[2], b_h[3], b_l[3];
#pragma unroll
            for (int mt = 0; mt < 2; ++mt) {
                a_h[mt] = *(const v16h*)&Ah[(((ch*2 + mt) * 32) + lane) * 16];
                a_l[mt] = *(const v16h*)&Al[(((ch*2 + mt) * 32) + lane) * 16];
            }
#pragma unroll
            for (int nt = 0; nt < 3; ++nt) {
                b_h[nt] = *(const v16h*)&Bh[((ch*3 + nt) * 32 + lane) * 16];
                b_l[nt] = *(const v16h*)&Bl[((ch*3 + nt) * 32 + lane) * 16];
            }
#pragma unroll
            for (int mt = 0; mt < 2; ++mt) {
#pragma unroll
                for (int nt = 0; nt < 3; ++nt) {
                    v8f c = acc[c2][mt][nt];
                    c = __builtin_amdgcn_wmma_f32_16x16x32_f16(false, a_l[mt], false, b_h[nt], (short)0, c, false, false);
                    c = __builtin_amdgcn_wmma_f32_16x16x32_f16(false, a_h[mt], false, b_l[nt], (short)0, c, false, false);
                    c = __builtin_amdgcn_wmma_f32_16x16x32_f16(false, a_h[mt], false, b_h[nt], (short)0, c, false, false);
                    acc[c2][mt][nt] = c;
                }
            }
        }
        __syncthreads();
    }

    // ---- epilogue: acc + bias -> LDS transpose buffer [m][p][c] (pad 17) ----
#pragma unroll
    for (int c2 = 0; c2 < 2; ++c2) {
        const int ch = wave * 2 + c2;
#pragma unroll
        for (int nt = 0; nt < 3; ++nt) {
            const int p  = nt * 16 + (lane & 15);
            const float bb = bias[(size_t)(c0 + ch) * Pn + p0 + p];
#pragma unroll
            for (int mt = 0; mt < 2; ++mt) {
#pragma unroll
                for (int v = 0; v < 8; ++v) {
                    const int m = mt * 16 + v + ((lane >> 4) << 3);
                    Fo[(m * 48 + p) * 17 + ch] = acc[c2][mt][nt][v] + bb;
                }
            }
        }
    }
    __syncthreads();

    // ---- coalesced store: 16 consecutive c per 16 lanes, + seq_last ----
#pragma unroll
    for (int j = 0; j < 2; ++j) {
        const int   m   = 2*q + j;
        const float slv = j ? sl1 : sl0;
        float* op = out + (size_t)(b0 + m) * (Pn*Cn) + (size_t)p0 * Cn + c0 + cA;
#pragma unroll 8
        for (int p = 0; p < 48; ++p) {
            op[p * Cn] = Fo[(m * 48 + p) * 17 + cA] + slv;
        }
    }
}

extern "C" void kernel_launch(void* const* d_in, const int* in_sizes, int n_in,
                              void* d_out, int out_size, void* d_ws, size_t ws_size,
                              hipStream_t stream) {
    const float* x    = (const float*)d_in[0];
    const float* w    = (const float*)d_in[1];
    const float* bias = (const float*)d_in[2];
    float* out = (float*)d_out;

    dim3 grid(Pn / 48, Bn / 32, Cn / 16);   // 7 x 128 x 4
    nlinear_wmma_f32split<<<grid, 256, 0, stream>>>(x, w, bias, out);
}
